// TransformerBlock_28724741275988
// MI455X (gfx1250) — compile-verified, matrix operand order checked by reading, not run on hardware
//
#include <hip/hip_runtime.h>
#include <hip/hip_bf16.h>

// ---------------------------------------------------------------------------
// Types for CDNA5 WMMA (wave32). 16x16x32 bf16 -> f32 accumulate.
// ---------------------------------------------------------------------------
typedef __attribute__((ext_vector_type(8)))  __bf16 v8bf;
typedef __attribute__((ext_vector_type(16))) __bf16 v16bf;
typedef __attribute__((ext_vector_type(8)))  float  v8f;
typedef __attribute__((ext_vector_type(4)))  unsigned int u32x4;
typedef __attribute__((ext_vector_type(8)))  int    i32x8;
typedef __attribute__((ext_vector_type(4)))  int    i32x4;

#define DIMC    2048
#define SEQ     2048
#define NHEADS  16
#define HDIM    128
#define HIDDENC 8192

// ---- bf16 <-> f32 helpers (bit-exact RNE, no reliance on native casts) ----
static __device__ __forceinline__ __bf16 f2bf(float f) {
    unsigned u; __builtin_memcpy(&u, &f, 4);
    unsigned r = (u + 0x7FFFu + ((u >> 16) & 1u)) >> 16;
    unsigned short h = (unsigned short)r;
    __bf16 b; __builtin_memcpy(&b, &h, 2);
    return b;
}
static __device__ __forceinline__ float bf2f(__bf16 b) {
    unsigned short h; __builtin_memcpy(&h, &b, 2);
    unsigned u = ((unsigned)h) << 16;
    float f; __builtin_memcpy(&f, &u, 4);
    return f;
}

static __device__ __forceinline__ v8f wmma_bf16(v16bf a, v16bf b, v8f c) {
    // (neg_a, A, neg_b, B, c_mod, C, reuse_a, reuse_b)
    return __builtin_amdgcn_wmma_f32_16x16x32_bf16(false, a, false, b,
                                                   (short)0, c, false, false);
}

// A/B fragment loader (16x32 bf16, M-major source with row stride `stride`):
// lane L (0..15) holds row M=L, K = 0..7 / 16..23; lane L+16: K = 8..15 / 24..31.
static __device__ __forceinline__ v16bf load_frag(const __bf16* base, int stride, int lane) {
    int r  = lane & 15;
    int kb = (lane >> 4) << 3;          // 0 or 8
    const __bf16* p = base + r * stride + kb;
    v8bf lo = *(const v8bf*)p;          // K = kb .. kb+7
    v8bf hi = *(const v8bf*)(p + 16);   // K = kb+16 .. kb+23
    return __builtin_shufflevector(lo, hi, 0,1,2,3,4,5,6,7,8,9,10,11,12,13,14,15);
}

// ---------------------------------------------------------------------------
// TDM: DMA a (tile_h x tile_w_dw dwords) tile of a 2D dword tensor into LDS.
// D# built per ISA ch.8: group0 = {flags, lds_addr, global_addr, type=2},
// group1 = {data_size=4B, tensor dims, tile dims, dim0 stride}.
// Descriptors are wave-uniform -> SGPRs. Tracked by TENSORcnt.
// ---------------------------------------------------------------------------
static __device__ __forceinline__ void tdm_load_tile(
    unsigned lds_addr, const void* gptr,
    unsigned tile_w_dw, unsigned tile_h,
    unsigned tensor_w_dw, unsigned tensor_h, unsigned long long row_stride_dw)
{
    unsigned long long ga = (unsigned long long)(size_t)gptr;
    u32x4 g0;
    g0.x = 1u;                                        // count=1, user mode
    g0.y = lds_addr;                                  // lds_addr[31:0]
    g0.z = (unsigned)(ga & 0xFFFFFFFFu);              // global_addr[31:0]
    g0.w = (unsigned)((ga >> 32) & 0x01FFFFFFu) | 0x80000000u; // addr[56:32] | type=2

    i32x8 g1;
    g1[0] = (int)(2u << 16);                          // data_size = 2 -> 4 bytes
    g1[1] = (int)((tensor_w_dw & 0xFFFFu) << 16);     // tensor_dim0[15:0] @ bits 63:48
    g1[2] = (int)(((tensor_w_dw >> 16) & 0xFFFFu) |   // tensor_dim0[31:16]
                  ((tensor_h & 0xFFFFu) << 16));      // tensor_dim1[15:0]
    g1[3] = (int)(((tensor_h >> 16) & 0xFFFFu) |      // tensor_dim1[31:16]
                  (tile_w_dw << 16));                 // tile_dim0
    g1[4] = (int)(tile_h & 0xFFFFu);                  // tile_dim1 (tile_dim2=0)
    g1[5] = (int)(unsigned)(row_stride_dw & 0xFFFFFFFFull);   // dim0_stride lo
    g1[6] = (int)(unsigned)((row_stride_dw >> 32) & 0xFFFFull); // dim0_stride hi
    g1[7] = 0;                                        // dim1_stride (2D: unused)
    i32x4 z4 = {0, 0, 0, 0};
#if defined(__clang_major__) && (__clang_major__ >= 23)
    i32x8 z8 = {0, 0, 0, 0, 0, 0, 0, 0};
    __builtin_amdgcn_tensor_load_to_lds(g0, g1, z4, z4, z8, 0);
#else
    __builtin_amdgcn_tensor_load_to_lds(g0, g1, z4, z4, 0);
#endif
}

// ---------------------------------------------------------------------------
// Weight convert + transpose: in[K][N] f32 -> out[N][K] bf16
// ---------------------------------------------------------------------------
__global__ void transpose_bf16_kernel(const float* __restrict__ in,
                                      __bf16* __restrict__ out, int K, int N) {
    int idx = blockIdx.x * 256 + threadIdx.x;
    if (idx >= K * N) return;
    int n = idx / K;
    int k = idx - n * K;
    out[idx] = f2bf(in[k * N + n]);
}

// ---------------------------------------------------------------------------
// RMSNorm: x[S][D] f32 -> out[S][D] bf16
// ---------------------------------------------------------------------------
__global__ __launch_bounds__(256) void rmsnorm_kernel(const float* __restrict__ x,
                                                      const float* __restrict__ w,
                                                      __bf16* __restrict__ out) {
    int row = blockIdx.x, tid = threadIdx.x;
    const float* xr = x + (long)row * DIMC;
    float ss = 0.f;
    for (int i = tid; i < DIMC; i += 256) { float v = xr[i]; ss += v * v; }
    __shared__ float red[256];
    red[tid] = ss; __syncthreads();
    for (int off = 128; off > 0; off >>= 1) {
        if (tid < off) red[tid] += red[tid + off];
        __syncthreads();
    }
    float inv = rsqrtf(red[0] / (float)DIMC + 1e-6f);
    for (int i = tid; i < DIMC; i += 256)
        out[(long)row * DIMC + i] = f2bf(xr[i] * inv * w[i]);
}

// ---------------------------------------------------------------------------
// TDM-fed, double-buffered WMMA GEMM: C[M][N] = A[M][K](bf16) * Bt[N][K](bf16)
// Block tile 128x128, BK=32, 8 waves, each wave a 64x32 subtile (4x2 WMMAs).
// Wave 0 drives the Tensor Data Mover; everyone else just computes.
// ---------------------------------------------------------------------------
static __device__ __forceinline__ void gemm_issue_tiles(
    const __bf16* A, const __bf16* Bt, int bm, int bn, int M, int N, int K,
    unsigned aBase, unsigned bBase, int buf, int k0)
{
    tdm_load_tile(aBase + (unsigned)buf * 8192u, A + (size_t)bm * K + k0,
                  16u, 128u, (unsigned)(K >> 1), (unsigned)M,
                  (unsigned long long)(K >> 1));
    tdm_load_tile(bBase + (unsigned)buf * 8192u, Bt + (size_t)bn * K + k0,
                  16u, 128u, (unsigned)(K >> 1), (unsigned)N,
                  (unsigned long long)(K >> 1));
}

template <bool OUTF32, bool RES>
__global__ __launch_bounds__(256) void gemm_kernel(const __bf16* __restrict__ A,
                                                   const __bf16* __restrict__ Bt,
                                                   const float* __restrict__ res,
                                                   void* __restrict__ outv,
                                                   int M, int N, int K) {
    __shared__ __align__(16) __bf16 As[2][128 * 32];   // 2 x 8 KB
    __shared__ __align__(16) __bf16 Bs[2][128 * 32];   // 2 x 8 KB
    int tid  = threadIdx.x;
    int lane = tid & 31;
    int wave = tid >> 5;
    int wm = (wave >> 2) * 64;          // 0 / 64
    int wn = (wave & 3) * 32;           // 0..96
    int bm = blockIdx.y * 128;
    int bn = blockIdx.x * 128;

    unsigned aBase = (unsigned)(size_t)(void*)&As[0][0];
    unsigned bBase = (unsigned)(size_t)(void*)&Bs[0][0];

    v8f acc[4][2];
    for (int i = 0; i < 4; ++i)
        for (int j = 0; j < 2; ++j)
            acc[i][j] = (v8f)(0.0f);

    if (wave == 0)
        gemm_issue_tiles(A, Bt, bm, bn, M, N, K, aBase, bBase, 0, 0);

    int cur = 0;
    for (int k0 = 0; k0 < K; k0 += 32) {
        if (wave == 0) __builtin_amdgcn_s_wait_tensorcnt(0);  // cur buffers ready
        __syncthreads();
        if (wave == 0 && k0 + 32 < K)                         // DMA next tile while
            gemm_issue_tiles(A, Bt, bm, bn, M, N, K,          // the WGP does math
                             aBase, bBase, cur ^ 1, k0 + 32);

        v16bf af[4], bf_[2];
        for (int mi = 0; mi < 4; ++mi)
            af[mi] = load_frag(&As[cur][(wm + mi * 16) * 32], 32, lane);
        for (int ni = 0; ni < 2; ++ni)
            bf_[ni] = load_frag(&Bs[cur][(wn + ni * 16) * 32], 32, lane);
        for (int mi = 0; mi < 4; ++mi)
            for (int ni = 0; ni < 2; ++ni)
                acc[mi][ni] = wmma_bf16(af[mi], bf_[ni], acc[mi][ni]);

        cur ^= 1;
        // next iteration's top barrier orders these reads vs. the next DMA
    }

    // Epilogue. C layout: lane L -> col = L&15; VGPR e -> row = e + 8*(L>>4).
    int rsel = (lane >> 4) << 3;
    for (int mi = 0; mi < 4; ++mi)
        for (int ni = 0; ni < 2; ++ni)
            for (int e = 0; e < 8; ++e) {
                int row = bm + wm + mi * 16 + e + rsel;
                int col = bn + wn + ni * 16 + (lane & 15);
                float v = acc[mi][ni][e];
                if (RES) v += res[(long)row * N + col];
                if (OUTF32) ((float*)outv)[(long)row * N + col] = v;
                else        ((__bf16*)outv)[(long)row * N + col] = f2bf(v);
            }
}

// ---------------------------------------------------------------------------
// RoPE (interleaved pairs), in-place on bf16 [S][NH*HD]
// ---------------------------------------------------------------------------
__global__ void rope_kernel(__bf16* __restrict__ x,
                            const float* __restrict__ cosb,
                            const float* __restrict__ sinb) {
    int idx = blockIdx.x * 256 + threadIdx.x;     // S*NH*64
    int i = idx & 63;
    int h = (idx >> 6) & 15;
    int s = idx >> 10;
    float c  = cosb[s * 64 + i];
    float sn = sinb[s * 64 + i];
    __bf16* p = x + (long)s * DIMC + h * HDIM + i * 2;
    float xr = bf2f(p[0]), xi = bf2f(p[1]);
    p[0] = f2bf(xr * c - xi * sn);
    p[1] = f2bf(xr * sn + xi * c);
}

// V transpose per head: vt[(h*HD+d)*S + s] = v[s*DIM + h*HD + d]
__global__ void vtrans_kernel(const __bf16* __restrict__ v, __bf16* __restrict__ vt) {
    int idx = blockIdx.x * 256 + threadIdx.x;     // S*DIM
    int s  = idx & (SEQ - 1);
    int hd = idx >> 11;
    vt[idx] = v[(long)s * DIMC + hd];
}

// ---------------------------------------------------------------------------
// Flash attention (causal). Grid: (S/64, NHEADS), block 128 (4 waves).
// Each wave: 16 query rows, full 128-dim ctx accumulator (8 x v8f).
// K/V fragments preloaded in bursts so global_load_b128s clause together.
// ---------------------------------------------------------------------------
__global__ __launch_bounds__(128) void attn_kernel(const __bf16* __restrict__ q,
                                                   const __bf16* __restrict__ k,
                                                   const __bf16* __restrict__ vt,
                                                   __bf16* __restrict__ ctx) {
    int qb   = blockIdx.x;
    int h    = blockIdx.y;
    int tid  = threadIdx.x;
    int lane = tid & 31;
    int wave = tid >> 5;
    int q0   = qb * 64 + wave * 16;

    __shared__ __align__(16) float  Ssh[4][16][32];
    __shared__ __align__(16) __bf16 Psh[4][16][32];
    __shared__ float mSh[4][16], lSh[4][16], aSh[4][16];

    if (lane < 16) { mSh[wave][lane] = -1e30f; lSh[wave][lane] = 0.f; }

    v16bf qf[4];
    for (int kt = 0; kt < 4; ++kt)
        qf[kt] = load_frag(q + (long)q0 * DIMC + h * HDIM + kt * 32, DIMC, lane);

    v8f acc[8];
    for (int d = 0; d < 8; ++d) acc[d] = (v8f)(0.0f);

    const float scale = 0.08838834764831845f;     // 1/sqrt(128)
    int nkb = 2 * qb + 2;                         // key blocks of 32, causal
    int rsel = (lane >> 4) << 3;

    for (int kb = 0; kb < nkb; ++kb) {
        int kbase = kb * 32;
        // --- scores: S(16x32) = Q(16x128) * K^T; preload 4 frags per burst ---
        v8f sacc[2];
        sacc[0] = (v8f)(0.0f); sacc[1] = (v8f)(0.0f);
        for (int nt = 0; nt < 2; ++nt) {
            v16bf kf[4];
            for (int kt = 0; kt < 4; ++kt)
                kf[kt] = load_frag(k + (long)(kbase + nt * 16) * DIMC + h * HDIM + kt * 32,
                                   DIMC, lane);
            for (int kt = 0; kt < 4; ++kt)
                sacc[nt] = wmma_bf16(qf[kt], kf[kt], sacc[nt]);
        }
        __builtin_prefetch(k + (long)(kbase + 32) * DIMC + h * HDIM, 0, 0);
        __builtin_prefetch(vt + (long)h * HDIM * SEQ + kbase + 32, 0, 0);
        __syncthreads();   // Psh/Ssh/aSh from previous iteration fully consumed
        for (int nt = 0; nt < 2; ++nt)
            for (int e = 0; e < 8; ++e) {
                int row = e + rsel;
                int col = nt * 16 + (lane & 15);
                float sv = sacc[nt][e] * scale;
                if (kbase + col > q0 + row) sv = -1e9f;   // causal mask
                Ssh[wave][row][col] = sv;
            }
        __syncthreads();
        // --- V fragments for the first half: overlap loads with softmax ---
        v16bf vf0[4];
        for (int dt = 0; dt < 4; ++dt)
            vf0[dt] = load_frag(vt + (long)(h * HDIM + dt * 16) * SEQ + kbase, SEQ, lane);
        // --- online softmax (one row per lane 0..15) ---
        if (lane < 16) {
            int r = lane;
            float m0 = mSh[wave][r], mx = m0;
            for (int c = 0; c < 32; ++c) mx = fmaxf(mx, Ssh[wave][r][c]);
            float alpha = __expf(m0 - mx);
            float sum = 0.f;
            for (int c = 0; c < 32; ++c) {
                float p = __expf(Ssh[wave][r][c] - mx);
                sum += p;
                Psh[wave][r][c] = f2bf(p);
            }
            lSh[wave][r] = lSh[wave][r] * alpha + sum;
            mSh[wave][r] = mx;
            aSh[wave][r] = alpha;
        }
        __syncthreads();
        // --- rescale ctx, then ctx += P(16x32) * V(32x128) ---
        for (int d = 0; d < 8; ++d)
            for (int e = 0; e < 8; ++e)
                acc[d][e] *= aSh[wave][e + rsel];
        v16bf pf = load_frag(&Psh[wave][0][0], 32, lane);
        for (int dt = 0; dt < 4; ++dt)
            acc[dt] = wmma_bf16(pf, vf0[dt], acc[dt]);
        v16bf vf1[4];
        for (int dt = 0; dt < 4; ++dt)
            vf1[dt] = load_frag(vt + (long)(h * HDIM + (dt + 4) * 16) * SEQ + kbase, SEQ, lane);
        for (int dt = 0; dt < 4; ++dt)
            acc[dt + 4] = wmma_bf16(pf, vf1[dt], acc[dt + 4]);
    }
    __syncthreads();
    for (int dt = 0; dt < 8; ++dt)
        for (int e = 0; e < 8; ++e) {
            int row = e + rsel;
            int col = lane & 15;
            float inv = 1.0f / lSh[wave][row];
            ctx[(long)(q0 + row) * DIMC + h * HDIM + dt * 16 + col] = f2bf(acc[dt][e] * inv);
        }
}

// silu(g) * u, elementwise bf16
__global__ void silu_kernel(const __bf16* __restrict__ g, const __bf16* __restrict__ u,
                            __bf16* __restrict__ s, int n) {
    int idx = blockIdx.x * 256 + threadIdx.x;
    if (idx >= n) return;
    float gf = bf2f(g[idx]);
    float sf = gf / (1.0f + __expf(-gf));
    s[idx] = f2bf(sf * bf2f(u[idx]));
}

// ---------------------------------------------------------------------------
// Host launcher
// ---------------------------------------------------------------------------
extern "C" void kernel_launch(void* const* d_in, const int* in_sizes, int n_in,
                              void* d_out, int out_size, void* d_ws, size_t ws_size,
                              hipStream_t stream) {
    const float* x    = (const float*)d_in[0];
    const float* fcos = (const float*)d_in[1];
    const float* fsin = (const float*)d_in[2];
    // d_in[3] = mask (causal handled analytically)
    const float* wq = (const float*)d_in[4];
    const float* wk = (const float*)d_in[5];
    const float* wv = (const float*)d_in[6];
    const float* wo = (const float*)d_in[7];
    const float* w1 = (const float*)d_in[8];
    const float* w2 = (const float*)d_in[9];
    const float* w3 = (const float*)d_in[10];
    const float* attn_w = (const float*)d_in[11];
    const float* ffn_w  = (const float*)d_in[12];

    char* ws = (char*)d_ws;
    const size_t MB = 1ull << 20;
    __bf16* wqT = (__bf16*)(ws + 0 * MB);     //  8 MB  [2048][2048]
    __bf16* wkT = (__bf16*)(ws + 8 * MB);
    __bf16* wvT = (__bf16*)(ws + 16 * MB);
    __bf16* woT = (__bf16*)(ws + 24 * MB);
    __bf16* w1T = (__bf16*)(ws + 32 * MB);    // 32 MB  [8192][2048]
    __bf16* w3T = (__bf16*)(ws + 64 * MB);    // 32 MB
    __bf16* w2T = (__bf16*)(ws + 96 * MB);    // 32 MB  [2048][8192]
    __bf16* hn  = (__bf16*)(ws + 128 * MB);   //  8 MB
    __bf16* qb  = (__bf16*)(ws + 136 * MB);   //  8 MB
    __bf16* kb  = (__bf16*)(ws + 144 * MB);   //  8 MB
    __bf16* vb  = (__bf16*)(ws + 152 * MB);   //  8 MB
    __bf16* vtb = (__bf16*)(ws + 160 * MB);   //  8 MB
    __bf16* ctx = (__bf16*)(ws + 168 * MB);   //  8 MB
    float*  hb  = (float*) (ws + 176 * MB);   // 16 MB
    __bf16* fn  = (__bf16*)(ws + 192 * MB);   //  8 MB
    __bf16* gb  = (__bf16*)(ws + 200 * MB);   // 32 MB
    __bf16* ub  = (__bf16*)(ws + 232 * MB);   // 32 MB
    __bf16* sb  = (__bf16*)(ws + 264 * MB);   // 32 MB

    // 1) weights -> bf16, transposed to [N][K]
    transpose_bf16_kernel<<<16384, 256, 0, stream>>>(wq, wqT, DIMC, DIMC);
    transpose_bf16_kernel<<<16384, 256, 0, stream>>>(wk, wkT, DIMC, DIMC);
    transpose_bf16_kernel<<<16384, 256, 0, stream>>>(wv, wvT, DIMC, DIMC);
    transpose_bf16_kernel<<<16384, 256, 0, stream>>>(wo, woT, DIMC, DIMC);
    transpose_bf16_kernel<<<65536, 256, 0, stream>>>(w1, w1T, DIMC, HIDDENC);
    transpose_bf16_kernel<<<65536, 256, 0, stream>>>(w3, w3T, DIMC, HIDDENC);
    transpose_bf16_kernel<<<65536, 256, 0, stream>>>(w2, w2T, HIDDENC, DIMC);

    // 2) attn rmsnorm
    rmsnorm_kernel<<<SEQ, 256, 0, stream>>>(x, attn_w, hn);

    // 3) Q/K/V projections
    dim3 g2k(DIMC / 128, SEQ / 128);
    gemm_kernel<false, false><<<g2k, 256, 0, stream>>>(hn, wqT, nullptr, qb, SEQ, DIMC, DIMC);
    gemm_kernel<false, false><<<g2k, 256, 0, stream>>>(hn, wkT, nullptr, kb, SEQ, DIMC, DIMC);
    gemm_kernel<false, false><<<g2k, 256, 0, stream>>>(hn, wvT, nullptr, vb, SEQ, DIMC, DIMC);

    // 4) RoPE on q,k ; transpose v per head
    rope_kernel<<<8192, 256, 0, stream>>>(qb, fcos, fsin);
    rope_kernel<<<8192, 256, 0, stream>>>(kb, fcos, fsin);
    vtrans_kernel<<<16384, 256, 0, stream>>>(vb, vtb);

    // 5) flash attention
    attn_kernel<<<dim3(SEQ / 64, NHEADS), 128, 0, stream>>>(qb, kb, vtb, ctx);

    // 6) h = x + ctx @ wo
    gemm_kernel<true, true><<<g2k, 256, 0, stream>>>(ctx, woT, x, hb, SEQ, DIMC, DIMC);

    // 7) ffn rmsnorm
    rmsnorm_kernel<<<SEQ, 256, 0, stream>>>(hb, ffn_w, fn);

    // 8) g = fn@w1 ; u = fn@w3 ; s = silu(g)*u
    dim3 g8k(HIDDENC / 128, SEQ / 128);
    gemm_kernel<false, false><<<g8k, 256, 0, stream>>>(fn, w1T, nullptr, gb, SEQ, HIDDENC, DIMC);
    gemm_kernel<false, false><<<g8k, 256, 0, stream>>>(fn, w3T, nullptr, ub, SEQ, HIDDENC, DIMC);
    silu_kernel<<<65536, 256, 0, stream>>>(gb, ub, sb, SEQ * HIDDENC);

    // 9) out = h + s@w2   (fp32)
    gemm_kernel<true, true><<<g2k, 256, 0, stream>>>(sb, w2T, hb, d_out, SEQ, DIMC, HIDDENC);
}
